// MultiHeadAttention_35253091566223
// MI455X (gfx1250) — matrix operand order repaired and checked by reading, not compiled, not run on hardware
//
#include <hip/hip_runtime.h>
#include <hip/hip_bf16.h>

typedef __bf16 bf16_t;
typedef __attribute__((ext_vector_type(16))) __bf16 v16bf;
typedef __attribute__((ext_vector_type(8)))  __bf16 v8bf;
typedef __attribute__((ext_vector_type(8)))  float  v8f;

#define WMMA_BF16(A, B, C) \
    __builtin_amdgcn_wmma_f32_16x16x32_bf16(false, (A), false, (B), (short)0, (C), false, false)

static __device__ __forceinline__ v16bf cat8(v8bf lo, v8bf hi) {
    return __builtin_shufflevector(lo, hi, 0,1,2,3,4,5,6,7,8,9,10,11,12,13,14,15);
}

#define H_  1024
#define S_  2048
#define B_  4
#define NH_ 16
#define HD_ 64

// ---------------- Kernel 1: fp32 -> bf16 convert ----------------
__global__ void cvt_f32_bf16(const float* __restrict__ src, bf16_t* __restrict__ dst, int n) {
    int i = blockIdx.x * blockDim.x + threadIdx.x;
    if (i < n) dst[i] = (bf16_t)src[i];
}

// ---------------- Kernel 2: QKV projection (bf16 WMMA, f32 accum) ----------------
// grid: (BS/16, H/64, 3), block: 32 (1 wave). Wave computes a 16x64 output strip.
__global__ __launch_bounds__(32) void qkv_proj(
    const bf16_t* __restrict__ Xb, const bf16_t* __restrict__ Wb,
    const float* __restrict__ bq, const float* __restrict__ bk, const float* __restrict__ bvv,
    bf16_t* __restrict__ Qb, bf16_t* __restrict__ Kb, bf16_t* __restrict__ Vtb) {
    const int lane = threadIdx.x;
    const int half = lane >> 4, n16 = lane & 15;
    const int row0 = blockIdx.x * 16;
    const int col0 = blockIdx.y * 64;
    const int mat  = blockIdx.z;            // 0=Q 1=K 2=V
    const bf16_t* W = Wb + (size_t)mat * H_ * H_;
    const float* bias = (mat == 0) ? bq : (mat == 1) ? bk : bvv;

    v8f acc[4] = {};
    const bf16_t* xrow = Xb + (size_t)(row0 + n16) * H_;   // A row m = row0 + (lane&15)
    for (int k0 = 0; k0 < H_; k0 += 32) {
        v16bf A = *(const v16bf*)(xrow + k0 + half * 16);
#pragma unroll
        for (int t = 0; t < 4; ++t) {
            int n = col0 + t * 16 + n16;                   // B col n = lane&15
            v16bf B = *(const v16bf*)(W + (size_t)n * H_ + k0 + half * 16);
            acc[t] = WMMA_BF16(A, B, acc[t]);
        }
    }
#pragma unroll
    for (int t = 0; t < 4; ++t) {
        int n = col0 + t * 16 + n16;
        float bsv  = bias[n];
        int head = n >> 6, d = n & 63;
        if (mat == 2) {
            // V transposed: [bh][d][s]; element r -> s contiguous => one b128 store
            v8bf pack;
#pragma unroll
            for (int r = 0; r < 8; ++r) pack[r] = (bf16_t)(acc[t][r] + bsv);
            int m = row0 + half * 8;
            int b = m / S_, s = m % S_;
            *(v8bf*)(Vtb + (((size_t)(b * NH_ + head)) * HD_ + d) * S_ + s) = pack;
        } else {
            bf16_t* dst   = (mat == 0) ? Qb : Kb;
            float   scale = (mat == 0) ? 0.125f : 1.0f;    // fold 1/sqrt(HD) into Q
#pragma unroll
            for (int r = 0; r < 8; ++r) {
                int m = row0 + half * 8 + r;               // C layout: m = r + 8*half
                int b = m / S_, s = m % S_;
                dst[(((size_t)(b * NH_ + head)) * S_ + s) * HD_ + d] =
                    (bf16_t)((acc[t][r] + bsv) * scale);
            }
        }
    }
}

// ---------------- Kernel 3: transposed flash attention ----------------
// grid: (S/64, B*NH), block: 128 (4 waves). Each wave: 16 queries, online softmax over S keys.
// Computes S^T = K·Q^T and ctx^T = V^T·P : each lane owns one query column throughout.
__global__ __launch_bounds__(128) void attn(
    const bf16_t* __restrict__ Qb, const bf16_t* __restrict__ Kb,
    const bf16_t* __restrict__ Vtb, bf16_t* __restrict__ Ctxb) {
    const int lane = threadIdx.x & 31, wave = threadIdx.x >> 5;
    const int half = lane >> 4, n16 = lane & 15;
    const int bh = blockIdx.y;
    const int b = bh >> 4, head = bh & 15;
    const int q0 = (blockIdx.x * 4 + wave) * 16;

    const bf16_t* Qbase = Qb  + (size_t)bh * S_ * HD_;
    const bf16_t* Kbase = Kb  + (size_t)bh * S_ * HD_;
    const bf16_t* Vbase = Vtb + (size_t)bh * HD_ * S_;

    // B operand of S^T gemm: Q^T, lane column = query q0+n16, loaded once
    const bf16_t* qrow = Qbase + (size_t)(q0 + n16) * HD_;
    v16bf bQ0 = cat8(*(const v8bf*)(qrow + half * 8),      *(const v8bf*)(qrow + 16 + half * 8));   // d = 8h+i | 16+8h+i
    v16bf bQ1 = cat8(*(const v8bf*)(qrow + 32 + half * 8), *(const v8bf*)(qrow + 48 + half * 8));   // d = 32+8h+i | 48+8h+i

    v8f acc[4] = {};                                       // ctx^T tiles: d-tiles x 16 queries
    float mq = -3.0e38f, lq = 0.0f;

    for (int j0 = 0; j0 < S_; j0 += 32) {
        v8f s_lo = {}, s_hi = {};
        {   // keys j0..j0+15
            const bf16_t* kr = Kbase + (size_t)(j0 + n16) * HD_;
            v16bf A0 = cat8(*(const v8bf*)(kr + half * 8),      *(const v8bf*)(kr + 16 + half * 8));
            v16bf A1 = cat8(*(const v8bf*)(kr + 32 + half * 8), *(const v8bf*)(kr + 48 + half * 8));
            s_lo = WMMA_BF16(A0, bQ0, s_lo);
            s_lo = WMMA_BF16(A1, bQ1, s_lo);
        }
        {   // keys j0+16..j0+31
            const bf16_t* kr = Kbase + (size_t)(j0 + 16 + n16) * HD_;
            v16bf A0 = cat8(*(const v8bf*)(kr + half * 8),      *(const v8bf*)(kr + 16 + half * 8));
            v16bf A1 = cat8(*(const v8bf*)(kr + 32 + half * 8), *(const v8bf*)(kr + 48 + half * 8));
            s_hi = WMMA_BF16(A0, bQ0, s_hi);
            s_hi = WMMA_BF16(A1, bQ1, s_hi);
        }
        // --- online softmax: lane owns query column; keys live in elements + other half ---
        float vmax = s_lo[0];
#pragma unroll
        for (int r = 0; r < 8; ++r) { vmax = fmaxf(vmax, s_lo[r]); vmax = fmaxf(vmax, s_hi[r]); }
        vmax = fmaxf(vmax, __shfl_xor(vmax, 16, 32));
        float mnew  = fmaxf(mq, vmax);
        float alpha = __expf(mq - mnew);
        float p_lo[8], p_hi[8], vsum = 0.0f;
#pragma unroll
        for (int r = 0; r < 8; ++r) {
            p_lo[r] = __expf(s_lo[r] - mnew);
            p_hi[r] = __expf(s_hi[r] - mnew);
            vsum += p_lo[r] + p_hi[r];
        }
        vsum += __shfl_xor(vsum, 16, 32);
        lq = lq * alpha + vsum;
        mq = mnew;
        // --- build B operand P[key][q]: own half supplies 8 keys, xor-16 half the other 8 ---
        v16bf Bp;
#pragma unroll
        for (int r = 0; r < 8; ++r) {
            float o_lo = __shfl_xor(p_lo[r], 16, 32);
            float o_hi = __shfl_xor(p_hi[r], 16, 32);
            Bp[r]     = (bf16_t)(half ? o_hi    : p_lo[r]);   // keys 16h + r
            Bp[r + 8] = (bf16_t)(half ? p_hi[r] : o_lo);      // keys 16h + 8 + r
        }
        // --- ctx^T += V^T · P  (A rows = d, contiguous key loads from transposed V) ---
#pragma unroll
        for (int t = 0; t < 4; ++t) {
#pragma unroll
            for (int r = 0; r < 8; ++r) acc[t][r] *= alpha;   // per-query scale (lane-uniform)
            const bf16_t* vr = Vbase + (size_t)(t * 16 + n16) * S_ + j0;
            v16bf Av = *(const v16bf*)(vr + half * 16);
            acc[t] = WMMA_BF16(Av, Bp, acc[t]);
        }
    }
    // normalize and store ctx (un-transpose is free: d runs along elements -> contiguous)
    float rl = 1.0f / lq;
    int s = q0 + n16;
    bf16_t* crow = Ctxb + ((size_t)(b * S_ + s)) * H_ + head * HD_;
#pragma unroll
    for (int t = 0; t < 4; ++t) {
        v8bf pack;
#pragma unroll
        for (int r = 0; r < 8; ++r) pack[r] = (bf16_t)(acc[t][r] * rl);
        *(v8bf*)(crow + t * 16 + half * 8) = pack;
    }
}

// ---------------- Kernel 4: output projection + residual + LayerNorm ----------------
// grid: (BS/16), block: 256 (8 waves). Block owns a full 16x1024 row strip so LN can fuse.
__global__ __launch_bounds__(256) void out_proj_ln(
    const bf16_t* __restrict__ Ctxb, const bf16_t* __restrict__ Wb,
    const float* __restrict__ bo, const float* __restrict__ hidden,
    const float* __restrict__ gamma, const float* __restrict__ beta,
    float* __restrict__ out) {
    const int lane = threadIdx.x & 31, wave = threadIdx.x >> 5;
    const int half = lane >> 4, n16 = lane & 15;
    const int row0 = blockIdx.x * 16;
    const bf16_t* Wo = Wb + (size_t)3 * H_ * H_;

    v8f acc[8] = {};
    const bf16_t* crow = Ctxb + (size_t)(row0 + n16) * H_;
    for (int k0 = 0; k0 < H_; k0 += 32) {
        v16bf A = *(const v16bf*)(crow + k0 + half * 16);
#pragma unroll
        for (int t = 0; t < 8; ++t) {
            int n = wave * 128 + t * 16 + n16;
            v16bf B = *(const v16bf*)(Wo + (size_t)n * H_ + k0 + half * 16);
            acc[t] = WMMA_BF16(A, B, acc[t]);
        }
    }
    // bias + residual in-place; per-lane LN partials over this wave's 128 columns
    float ps[8] = {}, pq[8] = {};
#pragma unroll
    for (int t = 0; t < 8; ++t) {
        int n = wave * 128 + t * 16 + n16;
        float bsv = bo[n];
#pragma unroll
        for (int r = 0; r < 8; ++r) {
            int m = row0 + half * 8 + r;
            float v = acc[t][r] + bsv + hidden[(size_t)m * H_ + n];
            acc[t][r] = v;
            ps[r] += v;
            pq[r] += v * v;
        }
    }
#pragma unroll
    for (int mask = 1; mask <= 8; mask <<= 1) {
#pragma unroll
        for (int r = 0; r < 8; ++r) {
            ps[r] += __shfl_xor(ps[r], mask, 32);
            pq[r] += __shfl_xor(pq[r], mask, 32);
        }
    }
    __shared__ float ssum[8][16], ssq[8][16];
    if (n16 == 0) {
#pragma unroll
        for (int r = 0; r < 8; ++r) {
            ssum[wave][half * 8 + r] = ps[r];
            ssq [wave][half * 8 + r] = pq[r];
        }
    }
    __syncthreads();
    float mean[8], rstd[8];
#pragma unroll
    for (int r = 0; r < 8; ++r) {
        float ts = 0.f, tq = 0.f;
#pragma unroll
        for (int w2 = 0; w2 < 8; ++w2) { ts += ssum[w2][half * 8 + r]; tq += ssq[w2][half * 8 + r]; }
        float mu  = ts * (1.0f / (float)H_);
        float var = tq * (1.0f / (float)H_) - mu * mu;
        mean[r] = mu;
        rstd[r] = rsqrtf(var + 1e-12f);
    }
#pragma unroll
    for (int t = 0; t < 8; ++t) {
        int n = wave * 128 + t * 16 + n16;
        float g = gamma[n], bb = beta[n];
#pragma unroll
        for (int r = 0; r < 8; ++r) {
            int m = row0 + half * 8 + r;
            out[(size_t)m * H_ + n] = (acc[t][r] - mean[r]) * rstd[r] * g + bb;
        }
    }
}

extern "C" void kernel_launch(void* const* d_in, const int* in_sizes, int n_in,
                              void* d_out, int out_size, void* d_ws, size_t ws_size,
                              hipStream_t stream) {
    (void)in_sizes; (void)n_in; (void)out_size; (void)ws_size;
    const float* hidden = (const float*)d_in[0];
    const float* Wq = (const float*)d_in[1];
    const float* bq = (const float*)d_in[2];
    const float* Wk = (const float*)d_in[3];
    const float* bk = (const float*)d_in[4];
    const float* Wv = (const float*)d_in[5];
    const float* bv = (const float*)d_in[6];
    const float* Wo = (const float*)d_in[7];
    const float* bo = (const float*)d_in[8];
    const float* gm = (const float*)d_in[9];
    const float* bt = (const float*)d_in[10];

    char* w = (char*)d_ws;
    bf16_t* Xb   = (bf16_t*)w; w += (size_t)B_ * S_ * H_ * 2;   // 16 MB
    bf16_t* Wb   = (bf16_t*)w; w += (size_t)4 * H_ * H_ * 2;    // 8 MB (Wq,Wk,Wv,Wo)
    bf16_t* Qb   = (bf16_t*)w; w += (size_t)B_ * S_ * H_ * 2;
    bf16_t* Kb   = (bf16_t*)w; w += (size_t)B_ * S_ * H_ * 2;
    bf16_t* Vtb  = (bf16_t*)w; w += (size_t)B_ * S_ * H_ * 2;
    bf16_t* Ctxb = (bf16_t*)w; w += (size_t)B_ * S_ * H_ * 2;

    const int nh = B_ * S_ * H_;
    const int nw = H_ * H_;
    cvt_f32_bf16<<<(nh + 255) / 256, 256, 0, stream>>>(hidden, Xb, nh);
    cvt_f32_bf16<<<(nw + 255) / 256, 256, 0, stream>>>(Wq, Wb + 0 * (size_t)nw, nw);
    cvt_f32_bf16<<<(nw + 255) / 256, 256, 0, stream>>>(Wk, Wb + 1 * (size_t)nw, nw);
    cvt_f32_bf16<<<(nw + 255) / 256, 256, 0, stream>>>(Wv, Wb + 2 * (size_t)nw, nw);
    cvt_f32_bf16<<<(nw + 255) / 256, 256, 0, stream>>>(Wo, Wb + 3 * (size_t)nw, nw);

    qkv_proj<<<dim3(B_ * S_ / 16, H_ / 64, 3), 32, 0, stream>>>(
        Xb, Wb, bq, bk, bv, Qb, Kb, Vtb);
    attn<<<dim3(S_ / 64, B_ * NH_), 128, 0, stream>>>(Qb, Kb, Vtb, Ctxb);
    out_proj_ln<<<dim3(B_ * S_ / 16), 256, 0, stream>>>(
        Ctxb, Wb, bo, hidden, gm, bt, (float*)d_out);
}
